// VQProsodyEncoder_41308995453491
// MI455X (gfx1250) — matrix operand order repaired and checked by reading, not compiled, not run on hardware
//
#include <hip/hip_runtime.h>
#include <math.h>

// ---------------- types for WMMA (CDNA5 / gfx1250, wave32) ----------------
typedef __attribute__((ext_vector_type(16))) __bf16 v16bf;
typedef __attribute__((ext_vector_type(8)))  __bf16 v8bf;
typedef __attribute__((ext_vector_type(8)))  float  v8f;
typedef __attribute__((ext_vector_type(4)))  unsigned int u32x4;
typedef __attribute__((ext_vector_type(8)))  unsigned int u32x8;

union Frag16 { v16bf v; v8bf h[2]; };

#define CH    384
#define KW    5
#define BINS  1024
#define BATCH 8
#define T1    4096
#define T2    512
#define NT    32          // time tile per workgroup
#define XT    (NT + 4)    // +2 halo each side (K=5, 'same' pad)
#define CCHUNK 32         // bf16 WMMA K depth
#define NCHUNK 60         // KW * (CH/CCHUNK)

// ---------------------------------------------------------------------------
// TDM: issue one tensor_load_to_lds for a 2D bf16 tile [tile_d1=384][tile_d0=32]
// with row stride 384 elements.  D# packing per CDNA5 ISA ch.8.3/8.4.
// ---------------------------------------------------------------------------
__device__ __forceinline__ void tdm_load_w(unsigned lds_off, const __bf16* gsrc) {
    unsigned long long ga = (unsigned long long)(const void*)gsrc;
    u32x4 g0;
    g0[0] = 1u;                                           // count=1, user desc
    g0[1] = lds_off;                                      // lds_addr (bytes)
    g0[2] = (unsigned)ga;                                 // global_addr[31:0]
    g0[3] = (unsigned)((ga >> 32) & 0x01ffffffull)        // global_addr[56:32]
          | 0x80000000u;                                  // type=2 ("image")
    u32x8 g1;
    g1[0] = 1u << 16;          // workgroup_mask=0 | data_size=1 (2 bytes)
    g1[1] = (unsigned)CCHUNK << 16;   // tensor_dim0[15:0]=32 @ bits[79:48]
    g1[2] = (unsigned)CH << 16;       // tensor_dim0 hi=0 | tensor_dim1[15:0]=384
    g1[3] = (unsigned)CCHUNK << 16;   // tensor_dim1 hi=0 | tile_dim0=32
    g1[4] = (unsigned)CH;             // tile_dim1=384 | tile_dim2=0
    g1[5] = (unsigned)CH;             // tensor_dim0_stride[31:0]=384
    g1[6] = 0u;                       // stride hi | tensor_dim1_stride lo
    g1[7] = 0u;
    asm volatile("tensor_load_to_lds %0, %1" :: "s"(g0), "s"(g1) : "memory");
}

// ---------------------------------------------------------------------------
// prep kernels: f32 -> bf16 repacks
// ---------------------------------------------------------------------------
__global__ void prep_weights(const float* __restrict__ w, __bf16* __restrict__ wb,
                             long total) {
    // w:  [L][co][ci][K]  ->  wb: [L][k][co][ci]
    long idx = (long)blockIdx.x * blockDim.x + threadIdx.x;
    if (idx >= total) return;
    int ci  = (int)(idx % CH);
    int co  = (int)((idx / CH) % CH);
    int k   = (int)((idx / ((long)CH * CH)) % KW);
    int lyr = (int)(idx / ((long)CH * CH * KW));
    wb[idx] = (__bf16)w[((((long)lyr * CH + co) * CH + ci) * KW) + k];
}

__global__ void prep_mel(const float* __restrict__ mel, __bf16* __restrict__ xb) {
    // mel [b][c][t] -> xb [b][t][c]
    long idx = (long)blockIdx.x * blockDim.x + threadIdx.x;
    if (idx >= (long)BATCH * T1 * CH) return;
    int c = (int)(idx % CH);
    int t = (int)((idx / CH) % T1);
    int b = (int)(idx / ((long)CH * T1));
    xb[idx] = (__bf16)mel[((long)b * CH + c) * T1 + t];
}

__global__ void prep_codebook(const float* __restrict__ cb, __bf16* __restrict__ cbh,
                              float* __restrict__ cnorm) {
    int code = blockIdx.x * blockDim.x + threadIdx.x;
    if (code >= BINS) return;
    float s = 0.f;
    for (int c = 0; c < CH; ++c) {
        float v = cb[(long)code * CH + c];
        s += v * v;
        cbh[(long)code * CH + c] = (__bf16)v;
    }
    cnorm[code] = s;
}

__global__ void zero_loss(float* loss) { if (threadIdx.x == 0) *loss = 0.f; }

// ---------------------------------------------------------------------------
// fused conv1d(K=5,'same') + bias + channel-LayerNorm + exact GELU
// x in/out layout: [b][t][c] bf16.
// WMMA orientation: A = X (M=time 16 x K=ci 32), B = W^T (K=ci 32 x N=co 16).
// Weight chunks arrive via TDM (double-buffered); X tile via async-to-LDS.
// ---------------------------------------------------------------------------
__global__ __launch_bounds__(256)
void conv_ln_gelu(const __bf16* __restrict__ xin,
                  const __bf16* __restrict__ wb,   // [k][co][ci] for this layer
                  const float*  __restrict__ bias,
                  const float*  __restrict__ gam,
                  const float*  __restrict__ bet,
                  __bf16*       __restrict__ xout,
                  int T) {
    __shared__ __align__(16) char smem[76800];
    __bf16* XL  = (__bf16*)smem;                 // [XT][CH]  27648 B
    __bf16* WL0 = (__bf16*)(smem + 27648);       // [CH][32]  24576 B
    __bf16* WL1 = (__bf16*)(smem + 52224);       // [CH][32]  24576 B
    const int CLP = CH + 8;                      // padded f32 row for LN reuse

    const int tid  = threadIdx.x;
    const int wave = tid >> 5;
    const int lane = tid & 31;
    const int l    = lane & 15;
    const int hi   = lane >> 4;
    const int t0   = blockIdx.x * NT;
    const int b    = blockIdx.y;
    const unsigned lds_base = (unsigned)(unsigned long long)(void*)smem;

    // ---- stage X tile: async global->LDS, zero-fill halo rows outside [0,T) ----
    for (int i = tid; i < XT * CH / 8; i += 256) {
        int row  = i / (CH / 8);
        int col8 = i % (CH / 8);
        int t    = t0 + row - 2;
        if (t >= 0 && t < T) {
            unsigned loff = lds_base + (unsigned)((row * CH + col8 * 8) * 2);
            unsigned long long g =
                (unsigned long long)(const void*)(xin + ((long)b * T + t) * CH + col8 * 8);
            asm volatile("global_load_async_to_lds_b128 %0, %1, off"
                         :: "v"(loff), "v"(g) : "memory");
        } else {
            *(uint4*)(XL + row * CH + col8 * 8) = make_uint4(0u, 0u, 0u, 0u);
        }
    }
    // ---- first weight chunk via Tensor Data Mover ----
    if (wave == 0) tdm_load_w(lds_base + 27648u, wb);
    asm volatile("s_wait_asynccnt 0x0" ::: "memory");
    if (wave == 0) __builtin_amdgcn_s_wait_tensorcnt(0);
    __syncthreads();

    v8f acc[2][3];
#pragma unroll
    for (int mt = 0; mt < 2; ++mt)
#pragma unroll
        for (int j = 0; j < 3; ++j) acc[mt][j] = (v8f){0.f,0.f,0.f,0.f,0.f,0.f,0.f,0.f};

    // ---- software-pipelined K loop: 5 taps x 12 ci-chunks ----
    for (int q = 0; q < NCHUNK; ++q) {
        const int k  = q / (CH / CCHUNK);
        const int cc = q % (CH / CCHUNK);
        const __bf16* WL = (q & 1) ? WL1 : WL0;
        // prefetch next chunk into the other buffer while we compute
        if ((q + 1 < NCHUNK) && wave == 0) {
            int k2  = (q + 1) / (CH / CCHUNK);
            int cc2 = (q + 1) % (CH / CCHUNK);
            unsigned noff = (q & 1) ? 27648u : 52224u;
            tdm_load_w(lds_base + noff, wb + (long)k2 * CH * CH + cc2 * CCHUNK);
        }

        // A fragments: lane<16: M=l, K 0..15; lane>=16: K 16..31
        Frag16 afr[2];
#pragma unroll
        for (int mt = 0; mt < 2; ++mt) {
            const __bf16* p = XL + (mt * 16 + l + k) * CH + cc * CCHUNK + hi * 16;
            afr[mt].h[0] = *(const v8bf*)p;
            afr[mt].h[1] = *(const v8bf*)(p + 8);
        }
        // B fragments: lane<16: N=co, K 0..15; lane>=16: K 16..31
        Frag16 bfr[3];
#pragma unroll
        for (int j = 0; j < 3; ++j) {
            int co = (wave * 3 + j) * 16 + l;
            bfr[j].v = *(const v16bf*)(WL + co * CCHUNK + hi * 16);
        }
#pragma unroll
        for (int mt = 0; mt < 2; ++mt)
#pragma unroll
            for (int j = 0; j < 3; ++j)
                acc[mt][j] = __builtin_amdgcn_wmma_f32_16x16x32_bf16(
                    false, afr[mt].v, false, bfr[j].v,
                    (short)0, acc[mt][j], false, false);

        if (wave == 0) __builtin_amdgcn_s_wait_tensorcnt(0);
        __syncthreads();
    }

    // ---- dump acc (+bias) into LDS f32 [t][co] for LayerNorm ----
    float* CL = (float*)smem;   // [NT][CLP] = 50176 B (reuses staging space)
#pragma unroll
    for (int mt = 0; mt < 2; ++mt)
#pragma unroll
        for (int j = 0; j < 3; ++j) {
            int cobase = (wave * 3 + j) * 16;
#pragma unroll
            for (int r = 0; r < 8; ++r) {
                int t  = mt * 16 + r + 8 * hi;
                int co = cobase + l;
                CL[t * CLP + co] = acc[mt][j][r] + bias[co];
            }
        }
    __syncthreads();

    // ---- LN over channels + affine + exact GELU; wave handles 4 time rows ----
    for (int rr = 0; rr < 4; ++rr) {
        int t = wave + rr * 8;
        float s = 0.f, sq = 0.f, vals[12];
#pragma unroll
        for (int i = 0; i < 12; ++i) {
            float v = CL[t * CLP + lane + i * 32];
            vals[i] = v; s += v; sq += v * v;
        }
#pragma unroll
        for (int off = 16; off > 0; off >>= 1) {
            s  += __shfl_xor(s,  off, 32);
            sq += __shfl_xor(sq, off, 32);
        }
        float mu   = s * (1.f / CH);
        float var  = sq * (1.f / CH) - mu * mu;
        float rstd = rsqrtf(var + 1e-5f);
        __bf16* outp = xout + ((long)b * T + (t0 + t)) * CH;
#pragma unroll
        for (int i = 0; i < 12; ++i) {
            int   co = lane + i * 32;
            float v  = (vals[i] - mu) * rstd * gam[co] + bet[co];
            float g  = 0.5f * v * (1.f + erff(v * 0.70710678118f));
            outp[co] = (__bf16)g;
        }
    }
}

// ---------------------------------------------------------------------------
// MaxPool1d(8,8) on [b][t][c] layout
// ---------------------------------------------------------------------------
__global__ void maxpool8(const __bf16* __restrict__ in, __bf16* __restrict__ out) {
    long idx = (long)blockIdx.x * blockDim.x + threadIdx.x;
    if (idx >= (long)BATCH * T2 * CH) return;
    int c  = (int)(idx % CH);
    int tp = (int)((idx / CH) % T2);
    int b  = (int)(idx / ((long)CH * T2));
    float m = -3.4e38f;
    for (int j = 0; j < 8; ++j)
        m = fmaxf(m, (float)in[((long)b * T1 + tp * 8 + j) * CH + c]);
    out[idx] = (__bf16)m;
}

// ---------------------------------------------------------------------------
// VQ: scores = x . c^T via WMMA; argmin(||c||^2 - 2s); gather f32 codebook;
// commit loss. x: [4096][384] bf16 (== flat (B*N, D)).
// ---------------------------------------------------------------------------
__global__ __launch_bounds__(256)
void vq_kernel(const __bf16* __restrict__ x,
               const __bf16* __restrict__ cbh,
               const float*  __restrict__ cnorm,
               const float*  __restrict__ cbf,     // original f32 codebook
               float* __restrict__ quant,          // [B][CH][T2]
               float* __restrict__ loss,
               float* __restrict__ ind) {
    __shared__ __bf16 XL[16 * CH];        // 12 KB
    __shared__ float bestv[16][8];
    __shared__ int   besti[16][8];

    const int tid  = threadIdx.x;
    const int wave = tid >> 5;
    const int lane = tid & 31;
    const int l    = lane & 15;
    const int hi   = lane >> 4;
    const int row0 = blockIdx.x * 16;

    for (int i = tid; i < 16 * CH / 8; i += 256)
        *(uint4*)(XL + i * 8) = *(const uint4*)(x + (long)row0 * CH + i * 8);
    __syncthreads();

    float runv[8]; int runi[8];
#pragma unroll
    for (int r = 0; r < 8; ++r) { runv[r] = 3.4e38f; runi[r] = 0; }

    for (int it = 0; it < BINS / 128; ++it) {
        int code0 = it * 128 + wave * 16;
        v8f acc = (v8f){0.f,0.f,0.f,0.f,0.f,0.f,0.f,0.f};
#pragma unroll
        for (int cc = 0; cc < CH / CCHUNK; ++cc) {
            Frag16 af, bf;
            const __bf16* ap = XL + l * CH + cc * CCHUNK + hi * 16;
            af.h[0] = *(const v8bf*)ap;
            af.h[1] = *(const v8bf*)(ap + 8);
            const __bf16* bp = cbh + (long)(code0 + l) * CH + cc * CCHUNK + hi * 16;
            bf.v = *(const v16bf*)bp;
            acc = __builtin_amdgcn_wmma_f32_16x16x32_bf16(
                false, af.v, false, bf.v, (short)0, acc, false, false);
        }
        int   code = code0 + l;
        float cn   = cnorm[code];
#pragma unroll
        for (int r = 0; r < 8; ++r) {
            float d  = cn - 2.f * acc[r];
            int   ci = code;
#pragma unroll
            for (int off = 1; off < 16; off <<= 1) {
                float ov = __shfl_xor(d,  off, 32);
                int   oi = __shfl_xor(ci, off, 32);
                if (ov < d || (ov == d && oi < ci)) { d = ov; ci = oi; }
            }
            if (d < runv[r] || (d == runv[r] && ci < runi[r])) { runv[r] = d; runi[r] = ci; }
        }
    }
    if (l == 0) {
#pragma unroll
        for (int r = 0; r < 8; ++r) {
            bestv[r + 8 * hi][wave] = runv[r];
            besti[r + 8 * hi][wave] = runi[r];
        }
    }
    __syncthreads();

    if (tid < 16) {
        float bv = bestv[tid][0]; int bi = besti[tid][0];
        for (int w = 1; w < 8; ++w) {
            if (bestv[tid][w] < bv || (bestv[tid][w] == bv && besti[tid][w] < bi)) {
                bv = bestv[tid][w]; bi = besti[tid][w];
            }
        }
        besti[tid][0] = bi;
        int rowg = row0 + tid;
        ind[rowg] = (float)bi;       // embed_ind flat [b][n] == row order
    }
    __syncthreads();

    // quantize output + commit loss
    int rowl = tid & 15;
    int rowg = row0 + rowl;
    int bb   = rowg >> 9;            // / T2
    int n    = rowg & (T2 - 1);
    int bi   = besti[rowl][0];
    const float* q = cbf + (long)bi * CH;
    float ls = 0.f;
    for (int c = tid >> 4; c < CH; c += 16) {
        float qq = q[c];
        float xx = (float)XL[rowl * CH + c];
        float dd = qq - xx;
        ls += dd * dd;
        quant[((long)bb * CH + c) * T2 + n] = qq;
    }
#pragma unroll
    for (int off = 16; off > 0; off >>= 1) ls += __shfl_xor(ls, off, 32);
    if (lane == 0)
        atomicAdd(loss, ls * (0.25f / ((float)BATCH * T2 * CH)));
}

// ---------------------------------------------------------------------------
extern "C" void kernel_launch(void* const* d_in, const int* in_sizes, int n_in,
                              void* d_out, int out_size, void* d_ws, size_t ws_size,
                              hipStream_t stream) {
    (void)in_sizes; (void)n_in; (void)out_size; (void)ws_size;
    const float* mel     = (const float*)d_in[0];
    const float* conv_w  = (const float*)d_in[1];
    const float* conv_b  = (const float*)d_in[2];
    const float* ln_g    = (const float*)d_in[3];
    const float* ln_b    = (const float*)d_in[4];
    const float* conv2_w = (const float*)d_in[5];
    const float* conv2_b = (const float*)d_in[6];
    const float* ln2_g   = (const float*)d_in[7];
    const float* ln2_b   = (const float*)d_in[8];
    const float* cbf     = (const float*)d_in[9];

    char* ws = (char*)d_ws;
    const long WELEM = (long)5 * KW * CH * CH;          // per weight stack
    __bf16* wb1   = (__bf16*)(ws);
    __bf16* wb2   = (__bf16*)(ws + WELEM * 2);
    __bf16* cbh   = (__bf16*)(ws + WELEM * 4);
    float*  cnorm = (float*) (ws + WELEM * 4 + (long)BINS * CH * 2);
    char*   actb  = ws + WELEM * 4 + (long)BINS * CH * 2 + BINS * 4 + 256;
    const long ACT = (long)BATCH * T1 * CH * 2;
    __bf16* act0  = (__bf16*)actb;
    __bf16* act1  = (__bf16*)(actb + ACT);

    prep_weights<<<(int)((WELEM + 255) / 256), 256, 0, stream>>>(conv_w,  wb1, WELEM);
    prep_weights<<<(int)((WELEM + 255) / 256), 256, 0, stream>>>(conv2_w, wb2, WELEM);
    prep_mel<<<(int)(((long)BATCH * T1 * CH + 255) / 256), 256, 0, stream>>>(mel, act0);
    prep_codebook<<<(BINS + 255) / 256, 256, 0, stream>>>(cbf, cbh, cnorm);

    // 5 pre-pool blocks, T = 4096
    __bf16* cin = act0; __bf16* cout = act1;
    for (int lyr = 0; lyr < 5; ++lyr) {
        conv_ln_gelu<<<dim3(T1 / NT, BATCH), 256, 0, stream>>>(
            cin, wb1 + (long)lyr * KW * CH * CH,
            conv_b + lyr * CH, ln_g + lyr * CH, ln_b + lyr * CH, cout, T1);
        __bf16* t = cin; cin = cout; cout = t;
    }
    // cin now holds last pre-pool output (act1)
    maxpool8<<<(int)(((long)BATCH * T2 * CH + 255) / 256), 256, 0, stream>>>(cin, cout);
    // cout (act0) holds pooled [b][512][c]
    cin = cout; cout = (cin == act0) ? act1 : act0;
    for (int lyr = 0; lyr < 5; ++lyr) {
        conv_ln_gelu<<<dim3(T2 / NT, BATCH), 256, 0, stream>>>(
            cin, wb2 + (long)lyr * KW * CH * CH,
            conv2_b + lyr * CH, ln2_g + lyr * CH, ln2_b + lyr * CH, cout, T2);
        __bf16* t = cin; cin = cout; cout = t;
    }

    float* quant    = (float*)d_out;                       // B*CH*T2
    float* loss_ptr = quant + (long)BATCH * CH * T2;
    float* ind_ptr  = loss_ptr + 1;
    zero_loss<<<1, 32, 0, stream>>>(loss_ptr);
    vq_kernel<<<(BATCH * T2) / 16, 256, 0, stream>>>(
        cin, cbh, cnorm, cbf, quant, loss_ptr, ind_ptr);
}
